// RNNAgent_14242111553609
// MI455X (gfx1250) — matrix operand order repaired and checked by reading, not compiled, not run on hardware
//
#include <hip/hip_runtime.h>
#include <math.h>

#define BB 64
#define TT 256
#define II 512
#define HH 1024
#define OO 512

typedef __attribute__((ext_vector_type(16))) _Float16 v16h;
typedef __attribute__((ext_vector_type(8)))  _Float16 v8h;
typedef __attribute__((ext_vector_type(8)))  float    v8f;
typedef int v4i __attribute__((vector_size(4 * sizeof(int))));

union Frag16 { v16h v; v8h h[2]; };

// Async copy of one 16-byte chunk global -> LDS (CDNA5 GLOBAL_LOAD_ASYNC_TO_LDS_B128,
// tracked by ASYNCcnt). Builtin signature (from hipcc diagnostic):
//   (v4i __device__* gsrc, v4i __shared__* lds, int offset, int cpol)
__device__ __forceinline__ void async_copy_b128(const _Float16* g, _Float16* l) {
#if __has_builtin(__builtin_amdgcn_global_load_async_to_lds_b128)
  __builtin_amdgcn_global_load_async_to_lds_b128(
      (__attribute__((address_space(1))) v4i*)(unsigned long long)(size_t)g,
      (__attribute__((address_space(3))) v4i*)(unsigned)(size_t)l,
      0, 0);
#else
  asm volatile("global_load_async_to_lds_b128 %0, %1, off"
               :: "v"((unsigned)(size_t)l), "v"(g) : "memory");
#endif
}

__device__ __forceinline__ void wait_asynccnt0() {
#if __has_builtin(__builtin_amdgcn_s_wait_asynccnt)
  __builtin_amdgcn_s_wait_asynccnt(0);
#else
  asm volatile("s_wait_asynccnt 0x0" ::: "memory");
#endif
}

// ---------------------------------------------------------------- converts
__global__ void cvt_f32_f16(const float* __restrict__ src,
                            _Float16* __restrict__ dst, int n) {
  int i = blockIdx.x * blockDim.x + threadIdx.x;
  if (i < n) dst[i] = (_Float16)src[i];
}

// ---------------------------------------------------------------- WMMA GEMM
// C[M,N] = A[M,K](f16) * W[N,K](f16)^T + bias[N]   (f32 accumulate)
// One wave computes a 16x64 strip (4 accumulators). Block = 8 waves (256 thr).
// Fragment layouts used here (same contiguous k order for A and B):
//   A 16x32 f16 : lane m=l&15; half=l>>4; elems e=0..15 -> K = 16*half + e (two b128 loads)
//   B 32x16 f16 : lane n=l&15; elems e=0..15 -> K = 16*half + e (two b128)
//   C/D 16x16 f32: reg r -> row = 8*half + r, col = l&15
__global__ void __launch_bounds__(256)
gemm_bias_f16(const _Float16* __restrict__ A, const _Float16* __restrict__ W,
              const float* __restrict__ bias, float* __restrict__ C,
              int M, int N, int K) {
  const int lane = threadIdx.x & 31;
  const int wave = threadIdx.x >> 5;
  const int l16  = lane & 15;
  const int hf   = lane >> 4;
  const int m0 = (blockIdx.y * 8 + wave) * 16;
  const int n0 = blockIdx.x * 64;
  if (m0 >= M) return;

  const _Float16* arow = A + (size_t)(m0 + l16) * K;
  v8f acc[4] = {};
  for (int k0 = 0; k0 < K; k0 += 32) {
    Frag16 a;
    a.h[0] = *reinterpret_cast<const v8h*>(arow + k0 + 16 * hf);
    a.h[1] = *reinterpret_cast<const v8h*>(arow + k0 + 16 * hf + 8);
    if (k0 + 32 < K) __builtin_prefetch(arow + k0 + 32, 0, 3);  // global_prefetch_b8 (near)
#pragma unroll
    for (int j = 0; j < 4; ++j) {
      const _Float16* wrow =
          W + (size_t)(n0 + 16 * j + l16) * K + (k0 + 16 * hf);
      Frag16 b;
      b.h[0] = *reinterpret_cast<const v8h*>(wrow);
      b.h[1] = *reinterpret_cast<const v8h*>(wrow + 8);
      acc[j] = __builtin_amdgcn_wmma_f32_16x16x32_f16(
          false, a.v, false, b.v, (short)0, acc[j], false, false);
    }
  }
#pragma unroll
  for (int j = 0; j < 4; ++j) {
    const int n = n0 + 16 * j + l16;
    const float bn = bias[n];
#pragma unroll
    for (int r = 0; r < 8; ++r) {
      const int m = m0 + 8 * hf + r;
      C[(size_t)m * N + n] = acc[j][r] + bn;
    }
  }
}

// ---------------------------------------------------------------- RNN step
// h_new[64,1024] = tanh( hprev(f16)[64,1024] @ Whh(f16)[1024,1024]^T
//                        + xw_t(f32) + b_hh )
// One wave per 16x16 tile; a block's 8 waves share one batch-row tile
// (mtile = blockIdx.x>>3) so the A panel (16 x 1024 f16 = 32KB) is async-staged
// into LDS once per block (GLOBAL_LOAD_ASYNC_TO_LDS_B128 + s_wait_asynccnt),
// cutting 8x redundant L2 traffic on the sequential critical path. B operand
// (unique 16 columns of Whh per wave) streams from global/L2.
__global__ void __launch_bounds__(256)
rnn_step(const _Float16* __restrict__ hprev, long hs,
         const _Float16* __restrict__ Whh,
         const float* __restrict__ xwt, long xs,
         const float* __restrict__ bias,
         _Float16* __restrict__ hout, long os,
         float* __restrict__ hout32) {
  __shared__ _Float16 smA[16 * HH];       // 32 KB A panel
  const int lane = threadIdx.x & 31;
  const int wave = threadIdx.x >> 5;
  const int l16  = lane & 15;
  const int hf   = lane >> 4;
  const int m0 = (blockIdx.x >> 3) * 16;              // batch tile (shared by block)
  const int n0 = ((blockIdx.x & 7) * 8 + wave) * 16;  // hidden tile (per wave)

  // Cooperative async stage of A: 16 rows x 1024 halfs = 2048 b128 chunks.
  for (int c = threadIdx.x; c < 16 * (HH / 8); c += 256) {
    const int row = c >> 7;            // c / (HH/8)
    const int col = (c & 127) * 8;     // element offset in halfs
    async_copy_b128(hprev + (size_t)(m0 + row) * hs + col, &smA[row * HH + col]);
  }
  wait_asynccnt0();
  __syncthreads();

  const _Float16* arow = &smA[l16 * HH];
  const _Float16* wrow = Whh + (size_t)(n0 + l16) * HH + 16 * hf;
  v8f acc = {};
  for (int k0 = 0; k0 < HH; k0 += 32) {
    Frag16 a, b;
    a.h[0] = *reinterpret_cast<const v8h*>(arow + k0 + 16 * hf);       // ds_load_b128
    a.h[1] = *reinterpret_cast<const v8h*>(arow + k0 + 16 * hf + 8);
    b.h[0] = *reinterpret_cast<const v8h*>(wrow + k0);
    b.h[1] = *reinterpret_cast<const v8h*>(wrow + k0 + 8);
    if (k0 + 32 < HH) __builtin_prefetch(wrow + k0 + 32, 0, 3);
    acc = __builtin_amdgcn_wmma_f32_16x16x32_f16(
        false, a.v, false, b.v, (short)0, acc, false, false);
  }
  const int n = n0 + l16;
  const float bn = bias[n];
#pragma unroll
  for (int r = 0; r < 8; ++r) {
    const int m = m0 + 8 * hf + r;
    float v = tanhf(acc[r] + xwt[(size_t)m * xs + n] + bn);
    hout[(size_t)m * os + n] = (_Float16)v;
    if (hout32) hout32[(size_t)m * HH + n] = v;
  }
}

// ---------------------------------------------------------------- softmax
__global__ void __launch_bounds__(256)
softmax_rows(const float* __restrict__ logits, float* __restrict__ out) {
  const int b = blockIdx.x;
  const int tid = threadIdx.x;
  __shared__ float red[256];
  float m = -INFINITY;
  for (int o = tid; o < OO; o += 256) m = fmaxf(m, logits[b * OO + o]);
  red[tid] = m; __syncthreads();
  for (int s = 128; s > 0; s >>= 1) {
    if (tid < s) red[tid] = fmaxf(red[tid], red[tid + s]);
    __syncthreads();
  }
  m = red[0]; __syncthreads();
  float sum = 0.f;
  for (int o = tid; o < OO; o += 256) sum += __expf(logits[b * OO + o] - m);
  red[tid] = sum; __syncthreads();
  for (int s = 128; s > 0; s >>= 1) {
    if (tid < s) red[tid] += red[tid + s];
    __syncthreads();
  }
  const float inv = 1.0f / red[0];
  for (int o = tid; o < OO; o += 256)
    out[b * OO + o] = __expf(logits[b * OO + o] - m) * inv;
}

// ---------------------------------------------------------------- launcher
extern "C" void kernel_launch(void* const* d_in, const int* in_sizes, int n_in,
                              void* d_out, int out_size, void* d_ws, size_t ws_size,
                              hipStream_t stream) {
  (void)in_sizes; (void)n_in; (void)out_size; (void)ws_size;
  const float* x    = (const float*)d_in[0];
  const float* h0   = (const float*)d_in[1];
  const float* Wih1 = (const float*)d_in[2];
  const float* Whh1 = (const float*)d_in[3];
  const float* bih1 = (const float*)d_in[4];
  const float* bhh1 = (const float*)d_in[5];
  const float* Wih2 = (const float*)d_in[6];
  const float* Whh2 = (const float*)d_in[7];
  const float* bih2 = (const float*)d_in[8];
  const float* bhh2 = (const float*)d_in[9];
  const float* Wfc  = (const float*)d_in[10];
  const float* bfc  = (const float*)d_in[11];
  float* out = (float*)d_out;

  // workspace carve (256B aligned)
  char* p = (char*)d_ws;
  auto carve = [&](size_t bytes) -> char* {
    char* r = p; p += (bytes + 255) & ~(size_t)255; return r;
  };
  _Float16* Wih1h = (_Float16*)carve((size_t)HH * II * 2);
  _Float16* Whh1h = (_Float16*)carve((size_t)HH * HH * 2);
  _Float16* Wih2h = (_Float16*)carve((size_t)HH * HH * 2);
  _Float16* Whh2h = (_Float16*)carve((size_t)HH * HH * 2);
  _Float16* Wfch  = (_Float16*)carve((size_t)OO * HH * 2);
  _Float16* xh    = (_Float16*)carve((size_t)BB * TT * II * 2);
  _Float16* h0h   = (_Float16*)carve((size_t)BB * HH * 2);
  _Float16* out1h = (_Float16*)carve((size_t)BB * TT * HH * 2);
  float*    xw    = (float*)   carve((size_t)BB * TT * HH * 4); // xw1 then xw2
  _Float16* h2a   = (_Float16*)carve((size_t)BB * HH * 2);
  _Float16* h2b   = (_Float16*)carve((size_t)BB * HH * 2);
  float*    logits= (float*)   carve((size_t)BB * OO * 4);

  auto cvt = [&](const float* s, _Float16* d, int n) {
    cvt_f32_f16<<<(n + 255) / 256, 256, 0, stream>>>(s, d, n);
  };
  cvt(x,    xh,    BB * TT * II);
  cvt(h0,   h0h,   BB * HH);
  cvt(Wih1, Wih1h, HH * II);
  cvt(Whh1, Whh1h, HH * HH);
  cvt(Wih2, Wih2h, HH * HH);
  cvt(Whh2, Whh2h, HH * HH);
  cvt(Wfc,  Wfch,  OO * HH);

  // layer-1 input projection: xw1[B*T, H] = x @ W_ih1^T + b_ih1
  {
    dim3 g(HH / 64, (BB * TT + 127) / 128);
    gemm_bias_f16<<<g, 256, 0, stream>>>(xh, Wih1h, bih1, xw,
                                         BB * TT, HH, II);
  }
  // layer-1 recurrence; hidden states stored into out1h[:,t,:] (f16)
  for (int t = 0; t < TT; ++t) {
    const _Float16* hp = (t == 0) ? h0h : (out1h + (size_t)(t - 1) * HH);
    long hs = (t == 0) ? (long)HH : (long)TT * HH;
    rnn_step<<<32, 256, 0, stream>>>(hp, hs, Whh1h,
                                     xw + (size_t)t * HH, (long)TT * HH,
                                     bhh1,
                                     out1h + (size_t)t * HH, (long)TT * HH,
                                     nullptr);
  }
  // layer-2 input projection: xw2 = out1 @ W_ih2^T + b_ih2 (reuse xw buffer)
  {
    dim3 g(HH / 64, (BB * TT + 127) / 128);
    gemm_bias_f16<<<g, 256, 0, stream>>>(out1h, Wih2h, bih2, xw,
                                         BB * TT, HH, HH);
  }
  // layer-2 recurrence; initial hidden = layer-1 final hidden = out1h[:,T-1,:]
  float* h2_f32_out = out + (size_t)BB * OO;  // [B,H] tail of d_out
  for (int t = 0; t < TT; ++t) {
    const _Float16* hp;
    long hs;
    if (t == 0) { hp = out1h + (size_t)(TT - 1) * HH; hs = (long)TT * HH; }
    else        { hp = (t & 1) ? h2a : h2b;           hs = (long)HH; }
    _Float16* ho = (t & 1) ? h2b : h2a;
    rnn_step<<<32, 256, 0, stream>>>(hp, hs, Whh2h,
                                     xw + (size_t)t * HH, (long)TT * HH,
                                     bhh2, ho, (long)HH,
                                     (t == TT - 1) ? h2_f32_out : nullptr);
  }
  // FC: logits[B,O] = h2 @ W_fc^T + b_fc   (T-1 = 255 odd -> final hidden in h2b)
  {
    dim3 g(OO / 64, (BB + 127) / 128);
    gemm_bias_f16<<<g, 256, 0, stream>>>(h2b, Wfch, bfc, logits, BB, OO, HH);
  }
  softmax_rows<<<BB, 256, 0, stream>>>(logits, out);
}
